// MultiHeadSelfAttention_70660801953827
// MI455X (gfx1250) — matrix operand order repaired and checked by reading, not compiled, not run on hardware
//
#include <hip/hip_runtime.h>

typedef float  v4f   __attribute__((ext_vector_type(4)));
typedef float  v8f   __attribute__((ext_vector_type(8)));
typedef __bf16 v4bf  __attribute__((ext_vector_type(4)));
typedef __bf16 v8bf  __attribute__((ext_vector_type(8)));
typedef __bf16 v16bf __attribute__((ext_vector_type(16)));

static __device__ __forceinline__ __bf16 f2bf(float f) {
    unsigned u = __builtin_bit_cast(unsigned, f);
    u += 0x7fffu + ((u >> 16) & 1u);           // round-to-nearest-even
    unsigned short h = (unsigned short)(u >> 16);
    return __builtin_bit_cast(__bf16, h);
}
static __device__ __forceinline__ float bf2f(__bf16 b) {
    unsigned short h = __builtin_bit_cast(unsigned short, b);
    unsigned u = ((unsigned)h) << 16;
    return __builtin_bit_cast(float, u);
}

// gfx1250 async copy: global -> LDS, 16 bytes per lane, tracked by ASYNCcnt.
// LDS operand is the wave-relative byte offset == low 32 bits of the generic
// shared-memory pointer.
static __device__ __forceinline__ void async_b128(const void* g, const void* l) {
    unsigned loff = (unsigned)(size_t)l;
    unsigned long long ga = (unsigned long long)(size_t)g;
    asm volatile("global_load_async_to_lds_b128 %0, %1, off"
                 :: "v"(loff), "v"(ga) : "memory");
}
static __device__ __forceinline__ void wait_async0() {
    asm volatile("s_wait_asynccnt 0" ::: "memory");
}

// ---------------------------------------------------------------------------
// One-shot f32 -> bf16 conversion (bandwidth bound; keeps conversion VALU work
// out of the GEMM hot loop).  n = 4096*4096, 8 elements/thread.
// ---------------------------------------------------------------------------
__global__ void __launch_bounds__(256)
cvt_f32_to_bf16(const float* __restrict__ s, __bf16* __restrict__ d)
{
    size_t i = ((size_t)blockIdx.x * 256 + threadIdx.x) * 8;
    v4f a = *(const v4f*)(s + i);
    v4f b = *(const v4f*)(s + i + 4);
    v8bf o;
    o[0] = f2bf(a[0]); o[1] = f2bf(a[1]); o[2] = f2bf(a[2]); o[3] = f2bf(a[3]);
    o[4] = f2bf(b[0]); o[5] = f2bf(b[1]); o[6] = f2bf(b[2]); o[7] = f2bf(b[3]);
    *(v8bf*)(d + i) = o;
}

// ---------------------------------------------------------------------------
// Projection GEMM, all-bf16 operands: out(4096x4096) = A @ W^T + bias.
// A row-major (m,k); W row-major (n,k) => NT GEMM, both k-contiguous.
// Block tile 128x128, K-tile 64, double-buffered LDS fed by async b128 copies.
// 8 waves: 2(M) x 4(N); each wave owns 4x2 WMMA 16x16 tiles.
// ---------------------------------------------------------------------------
template<bool SPLIT, bool OBF16>
__global__ void __launch_bounds__(256)
proj_gemm_bf16(const __bf16* __restrict__ A, const __bf16* __restrict__ W,
               const float* __restrict__ bias, void* __restrict__ outp)
{
    // row stride 72 bf16 = 144B keeps every 16B fragment read aligned
    __shared__ __attribute__((aligned(16))) __bf16 ldsA[2][128 * 72];
    __shared__ __attribute__((aligned(16))) __bf16 ldsB[2][128 * 72];

    const int tid  = threadIdx.x;
    const int lane = tid & 31;
    const int wv   = tid >> 5;
    const int mw   = wv & 1;          // 0..1
    const int nw   = wv >> 1;         // 0..3
    const int ln   = lane & 15;
    const int hi   = lane >> 4;       // lane half
    const int mBase = blockIdx.y * 128;
    const int nBase = blockIdx.x * 128;

    // 1024 16B-chunks per matrix per K-tile; 4 per thread
    int crow[4], ccol[4];
    #pragma unroll
    for (int c = 0; c < 4; ++c) {
        int ch = tid + c * 256;
        crow[c] = ch >> 3;
        ccol[c] = (ch & 7) * 8;
    }

    auto stage = [&](int buf, int k0) {
        #pragma unroll
        for (int c = 0; c < 4; ++c) {
            async_b128(A + (size_t)(mBase + crow[c]) * 4096 + k0 + ccol[c],
                       &ldsA[buf][crow[c] * 72 + ccol[c]]);
            async_b128(W + (size_t)(nBase + crow[c]) * 4096 + k0 + ccol[c],
                       &ldsB[buf][crow[c] * 72 + ccol[c]]);
        }
    };

    v8f acc[4][2] = {};

    stage(0, 0);
    wait_async0();
    __syncthreads();

    int cur = 0;
    for (int k0 = 0; k0 < 4096; k0 += 64) {
        const int nxt = cur ^ 1;
        if (k0 + 64 < 4096) stage(nxt, k0 + 64);   // overlap with compute

        #pragma unroll
        for (int ks = 0; ks < 64; ks += 32) {
            v16bf af[4];
            #pragma unroll
            for (int mt = 0; mt < 4; ++mt) {
                int m  = mw * 64 + mt * 16 + ln;
                int kb = ks + hi * 8;             // A frag: K {0..7,16..23}/{8..15,24..31}
                *((v8bf*)&af[mt])     = *(const v8bf*)(&ldsA[cur][m * 72 + kb]);
                *((v8bf*)&af[mt] + 1) = *(const v8bf*)(&ldsA[cur][m * 72 + kb + 16]);
            }
            #pragma unroll
            for (int nt = 0; nt < 2; ++nt) {
                int n  = nw * 32 + nt * 16 + ln;
                int kb = ks + hi * 8;             // B frag: K {0..7,16..23}/{8..15,24..31}
                v16bf bf;
                *((v8bf*)&bf)     = *(const v8bf*)(&ldsB[cur][n * 72 + kb]);
                *((v8bf*)&bf + 1) = *(const v8bf*)(&ldsB[cur][n * 72 + kb + 16]);
                #pragma unroll
                for (int mt = 0; mt < 4; ++mt)
                    acc[mt][nt] = __builtin_amdgcn_wmma_f32_16x16x32_bf16(
                        false, af[mt], false, bf, (short)0, acc[mt][nt], false, false);
            }
        }
        if (k0 + 64 < 4096) wait_async0();
        __syncthreads();
        cur = nxt;
    }

    // ---- epilogue: bias + store ----
    #pragma unroll
    for (int nt = 0; nt < 2; ++nt) {
        int col = nBase + nw * 32 + nt * 16 + ln;
        float bv = bias[col];
        #pragma unroll
        for (int mt = 0; mt < 4; ++mt) {
            #pragma unroll
            for (int r = 0; r < 8; ++r) {
                int row = mBase + mw * 64 + mt * 16 + r + hi * 8;
                float val = acc[mt][nt][r] + bv;
                size_t dst;
                if (SPLIT) {
                    // (b,s, h*64+d) -> (b,h,s,d): row = b*64+s, col = h*64+d
                    dst = (size_t)((row >> 6) * 64 + (col >> 6)) * 4096
                        + (size_t)((row & 63) * 64 + (col & 63));
                } else {
                    dst = (size_t)row * 4096 + col;
                }
                if (OBF16) ((__bf16*)outp)[dst] = f2bf(val);
                else       ((float*)outp)[dst]  = val;
            }
        }
    }
}

// ---------------------------------------------------------------------------
// RoPE (reference quirk): pairs along d, angle = h * 10000^(-(s*32+d/2)/2048)
// ---------------------------------------------------------------------------
__global__ void __launch_bounds__(256) rope_kernel(__bf16* __restrict__ t)
{
    size_t p = (size_t)blockIdx.x * 256 + threadIdx.x;   // pair index
    int d2 = (int)(p & 31);
    int s  = (int)((p >> 5) & 63);
    int h  = (int)((p >> 11) & 63);
    int m  = s * 32 + d2;
    float freq = __expf(-(float)m * 0.0044972365f);      // ln(10000)/2048
    float theta = (float)h * freq;
    float sn, cs;
    __sincosf(theta, &sn, &cs);
    size_t flat = p * 2;
    float re = bf2f(t[flat]);
    float im = bf2f(t[flat + 1]);
    t[flat]     = f2bf(re * cs - im * sn);
    t[flat + 1] = f2bf(re * sn + im * cs);
}

// ---------------------------------------------------------------------------
// Attention: one workgroup per (b,h); q,k staged via async copies, v staged
// transposed through VGPRs.  4 waves; wave w owns query rows [w*16, w*16+16).
// ---------------------------------------------------------------------------
__global__ void __launch_bounds__(128)
attention_kernel(const __bf16* __restrict__ q, const __bf16* __restrict__ k,
                 const __bf16* __restrict__ v, __bf16* __restrict__ attn)
{
    __shared__ __attribute__((aligned(16))) __bf16 qs[64 * 72];
    __shared__ __attribute__((aligned(16))) __bf16 ks[64 * 72];
    __shared__ __attribute__((aligned(16))) __bf16 vT[64 * 72];   // vT[d][j] = v[j][d]
    __shared__ __attribute__((aligned(16))) __bf16 ps[64 * 72];   // softmax probs
    __shared__ float sc[64 * 66];

    const int bh   = blockIdx.x;
    const int tid  = threadIdx.x;
    const int lane = tid & 31;
    const int w    = tid >> 5;
    const int ln   = lane & 15;
    const int hi   = lane >> 4;
    const size_t base = (size_t)bh * 4096;

    // stage q, k via async DMA; v transposed through VGPRs
    for (int c = tid; c < 512; c += 128) {
        int row  = c >> 3;
        int col8 = (c & 7) * 8;
        async_b128(q + base + row * 64 + col8, &qs[row * 72 + col8]);
        async_b128(k + base + row * 64 + col8, &ks[row * 72 + col8]);
        v8bf dv = *(const v8bf*)(v + base + row * 64 + col8);
        #pragma unroll
        for (int i = 0; i < 8; ++i) vT[(col8 + i) * 72 + row] = dv[i];
    }
    wait_async0();
    __syncthreads();

    // ---- scores = q @ k^T * (1/8) ----
    v8f acc[4] = {};
    #pragma unroll
    for (int kk = 0; kk < 64; kk += 32) {
        int am = w * 16 + ln;
        int ak = kk + hi * 8;
        v16bf af;
        *((v8bf*)&af)     = *(const v8bf*)(&qs[am * 72 + ak]);
        *((v8bf*)&af + 1) = *(const v8bf*)(&qs[am * 72 + ak + 16]);
        #pragma unroll
        for (int nt = 0; nt < 4; ++nt) {
            int bn = nt * 16 + ln;
            int bk = kk + hi * 8;
            v16bf bf;
            *((v8bf*)&bf)     = *(const v8bf*)(&ks[bn * 72 + bk]);
            *((v8bf*)&bf + 1) = *(const v8bf*)(&ks[bn * 72 + bk + 16]);
            acc[nt] = __builtin_amdgcn_wmma_f32_16x16x32_bf16(
                false, af, false, bf, (short)0, acc[nt], false, false);
        }
    }
    #pragma unroll
    for (int nt = 0; nt < 4; ++nt)
        #pragma unroll
        for (int r = 0; r < 8; ++r) {
            int row = w * 16 + r + hi * 8;
            int col = nt * 16 + ln;
            sc[row * 66 + col] = acc[nt][r] * 0.125f;
        }
    __syncthreads();

    // ---- softmax over each row ----
    if (tid < 64) {
        float mx = -1e30f;
        for (int j = 0; j < 64; ++j) mx = fmaxf(mx, sc[tid * 66 + j]);
        float sum = 0.f;
        for (int j = 0; j < 64; ++j) {
            float e = __expf(sc[tid * 66 + j] - mx);
            sc[tid * 66 + j] = e;
            sum += e;
        }
        float inv = 1.0f / sum;
        for (int j = 0; j < 64; ++j) ps[tid * 72 + j] = f2bf(sc[tid * 66 + j] * inv);
    }
    __syncthreads();

    // ---- out = P @ V ----
    v8f oacc[4] = {};
    #pragma unroll
    for (int kk = 0; kk < 64; kk += 32) {
        int am = w * 16 + ln;
        int ak = kk + hi * 8;
        v16bf af;
        *((v8bf*)&af)     = *(const v8bf*)(&ps[am * 72 + ak]);
        *((v8bf*)&af + 1) = *(const v8bf*)(&ps[am * 72 + ak + 16]);
        #pragma unroll
        for (int nt = 0; nt < 4; ++nt) {
            int bn = nt * 16 + ln;            // n = d
            int bk = kk + hi * 8;             // k = j
            v16bf bf;
            *((v8bf*)&bf)     = *(const v8bf*)(&vT[bn * 72 + bk]);
            *((v8bf*)&bf + 1) = *(const v8bf*)(&vT[bn * 72 + bk + 16]);
            oacc[nt] = __builtin_amdgcn_wmma_f32_16x16x32_bf16(
                false, af, false, bf, (short)0, oacc[nt], false, false);
        }
    }
    // store back as (B, S, h*64+d) bf16 for the o-projection
    const int b = bh >> 6, h = bh & 63;
    #pragma unroll
    for (int nt = 0; nt < 4; ++nt)
        #pragma unroll
        for (int r = 0; r < 8; ++r) {
            int s = w * 16 + r + hi * 8;
            int d = nt * 16 + ln;
            attn[(size_t)(b * 64 + s) * 4096 + h * 64 + d] = f2bf(oacc[nt][r]);
        }
}

// ---------------------------------------------------------------------------
extern "C" void kernel_launch(void* const* d_in, const int* in_sizes, int n_in,
                              void* d_out, int out_size, void* d_ws, size_t ws_size,
                              hipStream_t stream) {
    const float* x   = (const float*)d_in[0];
    const float* w_q = (const float*)d_in[1];
    const float* b_q = (const float*)d_in[2];
    const float* w_k = (const float*)d_in[3];
    const float* b_k = (const float*)d_in[4];
    const float* w_v = (const float*)d_in[5];
    const float* b_v = (const float*)d_in[6];
    const float* w_o = (const float*)d_in[7];
    const float* b_o = (const float*)d_in[8];
    float* out = (float*)d_out;

    const size_t TS = (size_t)4096 * 4096;
    __bf16* q    = (__bf16*)d_ws;        // ws usage: 5 * 32 MB = 160 MB
    __bf16* kbuf = q + TS;
    __bf16* vbuf = kbuf + TS;
    __bf16* xbf  = vbuf + TS;            // reused as attention output later
    __bf16* wbuf = xbf + TS;             // shared bf16 weight slot (stream-serialized)
    __bf16* attn = xbf;                  // alias: x-bf16 dead after v-projection

    dim3 grid(32, 32), blk(256);
    const int cvtBlocks = (int)(TS / 8 / 256);   // 8192

    cvt_f32_to_bf16<<<cvtBlocks, 256, 0, stream>>>(x, xbf);

    cvt_f32_to_bf16<<<cvtBlocks, 256, 0, stream>>>(w_q, wbuf);
    proj_gemm_bf16<true,  true ><<<grid, blk, 0, stream>>>(xbf, wbuf, b_q, q);

    cvt_f32_to_bf16<<<cvtBlocks, 256, 0, stream>>>(w_k, wbuf);
    proj_gemm_bf16<true,  true ><<<grid, blk, 0, stream>>>(xbf, wbuf, b_k, kbuf);

    cvt_f32_to_bf16<<<cvtBlocks, 256, 0, stream>>>(w_v, wbuf);
    proj_gemm_bf16<true,  true ><<<grid, blk, 0, stream>>>(xbf, wbuf, b_v, vbuf);

    rope_kernel<<<32768, 256, 0, stream>>>(q);
    rope_kernel<<<32768, 256, 0, stream>>>(kbuf);

    attention_kernel<<<4096, 128, 0, stream>>>(q, kbuf, vbuf, attn);

    cvt_f32_to_bf16<<<cvtBlocks, 256, 0, stream>>>(w_o, wbuf);
    proj_gemm_bf16<false, false><<<grid, blk, 0, stream>>>(attn, wbuf, b_o, out);
}
